// LDCAugmentation_84052509982728
// MI455X (gfx1250) — compile-verified, matrix operand order checked by reading, not run on hardware
//
#include <hip/hip_runtime.h>

typedef __attribute__((ext_vector_type(16))) __bf16 v16bf;
typedef __attribute__((ext_vector_type(8)))  __bf16 v8bf;
typedef __attribute__((ext_vector_type(8)))  float  v8f;

constexpr int N = 262144;
constexpr int M = 131072;
constexpr int C = 256;
constexpr int H = 32;
constexpr float EPS = 1e-5f;

// ---------------- workspace layout (bytes) ----------------
// rel (bf16 N*C) and x2 (bf16 N*C) alias: rel dies after k_gemm1, x2 born in k_fused2.
constexpr size_t REL_OFF = 0;
constexpr size_t X2_OFF  = 0;
constexpr size_t T_OFF   = (size_t)N * C * 2;              // f32 N*H
constexpr size_t PK_OFF  = T_OFF + (size_t)N * H * 4;      // packed bf16 weight frags
constexpr int NFRAG_W1 = (H / 16) * (C / 32);              // 16
constexpr int NFRAG_W2 = (C / 16) * (H / 32);              // 16
constexpr int NFRAG_WL = (C / 16) * (C / 32);              // 128
constexpr size_t W1P_H = 0;
constexpr size_t W2P_H = W1P_H + (size_t)NFRAG_W1 * 512;
constexpr size_t WLP_H = W2P_H + (size_t)NFRAG_W2 * 512;
constexpr size_t PK_HALVES = WLP_H + (size_t)NFRAG_WL * 512;   // 81920
constexpr size_t STAT_OFF = PK_OFF + PK_HALVES * 2;
// stat region float indices
constexpr int SUMA = 0,    SQA = 256,  SUMB = 512, SQB = 544;
constexpr int SUMM = 576,  SQM = 832;
constexpr int SA = 1088, BA = 1344, SB = 1600, BB = 1632, SM = 1664, BM = 1920;
constexpr int NSTAT_ZERO = 1088;

// CDNA5 16-bit A/B fragment: element e of lane L holds K index kmap(L,e)
__device__ __forceinline__ int kmap(int lane, int e) {
    return ((lane >> 4) << 3) + ((e >> 3) << 4) + (e & 7);
}

__device__ __forceinline__ v16bf cat16(v8bf lo, v8bf hi) {
    return __builtin_shufflevector(lo, hi, 0, 1, 2, 3, 4, 5, 6, 7,
                                   8, 9, 10, 11, 12, 13, 14, 15);
}

// ---------------- kernel 0: pack weights into B-fragment-linear bf16, zero stats ----------------
__global__ void k_prep(const float* __restrict__ W1, const float* __restrict__ W2,
                       const float* __restrict__ Wl, __bf16* __restrict__ pk,
                       float* __restrict__ st) {
    int i = blockIdx.x * 256 + threadIdx.x;
    if (i < (int)PK_HALVES) {
        int frag = i >> 9;
        int lane = (i >> 4) & 31;
        int e = i & 15;
        float v;
        if (frag < NFRAG_W1) {                       // W1: (H,C), B[k][n]=W1[n][k]
            int ct = frag >> 3, ks = frag & 7;
            int n = ct * 16 + (lane & 15);
            int k = ks * 32 + kmap(lane, e);
            v = W1[n * C + k];
        } else if (frag < NFRAG_W1 + NFRAG_W2) {     // W2: (C,H)
            int f = frag - NFRAG_W1;
            int n = f * 16 + (lane & 15);
            int k = kmap(lane, e);
            v = W2[n * H + k];
        } else {                                      // Wlin: (C,C)
            int f = frag - NFRAG_W1 - NFRAG_W2;
            int ct = f >> 3, ks = f & 7;
            int n = ct * 16 + (lane & 15);
            int k = ks * 32 + kmap(lane, e);
            v = Wl[n * C + k];
        }
        pk[i] = (__bf16)v;
    } else {
        int j = i - (int)PK_HALVES;
        if (j < NSTAT_ZERO) st[j] = 0.0f;
    }
}

// ---------------- kernel 1: rel = x - gx[idx] (bf16), per-channel sum/sumsq ----------------
__global__ void __launch_bounds__(256) k_rel_stats(
    const float* __restrict__ x, const float* __restrict__ gx,
    const long long* __restrict__ idx, __bf16* __restrict__ rel,
    float* __restrict__ sumA, float* __restrict__ sqA) {
    const int c = threadIdx.x;          // one channel per thread
    const int row0 = blockIdx.x * 32;
    float s = 0.f, q = 0.f;
    for (int r = 0; r < 32; ++r) {
        int row = row0 + r;
        long long g = idx[row];
        float aug = gx[(size_t)g * C + c];
        float v = x[(size_t)row * C + c] - aug;
        rel[(size_t)row * C + c] = (__bf16)v;
        s += v;
        q += v * v;
    }
    atomicAdd(&sumA[c], s);
    atomicAdd(&sqA[c], q);
}

// ---------------- finalize BN stats -> scale/shift ----------------
__global__ void k_finalize(const float* __restrict__ sum, const float* __restrict__ sq,
                           const float* __restrict__ g, const float* __restrict__ b,
                           float* __restrict__ scale, float* __restrict__ shift,
                           int n, float invN) {
    int c = blockIdx.x * blockDim.x + threadIdx.x;
    if (c >= n) return;
    float mu = sum[c] * invN;
    float var = sq[c] * invN - mu * mu;
    float s = g[c] * rsqrtf(var + EPS);
    scale[c] = s;
    shift[c] = b[c] - mu * s;
}

// ---------------- kernel 2: t = relu(bn(rel)) @ W1^T + b1 (WMMA), stats over H ----------------
__global__ void __launch_bounds__(256) k_gemm1(
    const __bf16* __restrict__ rel, const __bf16* __restrict__ pk,
    const float* __restrict__ st, const float* __restrict__ b1,
    float* __restrict__ t, float* __restrict__ sumB, float* __restrict__ sqB) {
    __shared__ __bf16 Abuf[2][64 * 40];   // double-buffered 64x32 tiles, stride 40
    __shared__ float sLoc[H], qLoc[H];
    const int tid = threadIdx.x, lane = tid & 31, wv = tid >> 5;
    const int rowTile = wv >> 1;          // 0..3
    const int colTile = wv & 1;           // 0..1
    const int row0 = blockIdx.x * 64;
    const float* sa = st + SA;
    const float* ba = st + BA;
    if (tid < H) { sLoc[tid] = 0.f; qLoc[tid] = 0.f; }

    const int sr = tid >> 2, sq_ = tid & 3;     // staging coords
    auto stage = [&](int ks) {
        int k0 = ks * 32 + sq_ * 8;
        v8bf rv = *(const v8bf*)(rel + (size_t)(row0 + sr) * C + k0);
        __bf16* dst = Abuf[ks & 1] + sr * 40 + sq_ * 8;
#pragma unroll
        for (int e = 0; e < 8; ++e) {
            float v = (float)rv[e] * sa[k0 + e] + ba[k0 + e];
            dst[e] = (__bf16)fmaxf(v, 0.0f);
        }
    };

    v8f acc = {};
    const int mloc = rowTile * 16 + (lane & 15);
    const int kbase = (lane >> 4) << 3;
    stage(0);
    __syncthreads();
    for (int ks = 0; ks < C / 32; ++ks) {
        const __bf16* buf = Abuf[ks & 1];
        v8bf lo = *(const v8bf*)(buf + mloc * 40 + kbase);
        v8bf hi = *(const v8bf*)(buf + mloc * 40 + kbase + 16);
        if (ks < C / 32 - 1) stage(ks + 1);     // overlap next stage with WMMA
        v16bf a = cat16(lo, hi);
        v16bf b = *(const v16bf*)(pk + W1P_H + ((size_t)(colTile * 8 + ks)) * 512 + lane * 16);
        acc = __builtin_amdgcn_wmma_f32_16x16x32_bf16(false, a, false, b,
                                                      (short)0, acc, false, false);
        __syncthreads();
    }
    const int n = colTile * 16 + (lane & 15);
    const float bias = b1[n];
    float s = 0.f, q2 = 0.f;
#pragma unroll
    for (int j = 0; j < 8; ++j) {
        int m = rowTile * 16 + j + ((lane >> 4) << 3);
        float v = acc[j] + bias;
        t[(size_t)(row0 + m) * H + n] = v;
        s += v;
        q2 += v * v;
    }
    atomicAdd(&sLoc[n], s);
    atomicAdd(&qLoc[n], q2);
    __syncthreads();
    if (tid < H) { atomicAdd(&sumB[tid], sLoc[tid]); atomicAdd(&sqB[tid], qLoc[tid]); }
}

// ---- kernel 3: fused bn+GEMM2+softmax+gather-combine, x2 store (bf16), GEMM3 stats ----
__global__ void __launch_bounds__(256) k_fused2(
    const float* __restrict__ t, const __bf16* __restrict__ pk,
    const float* __restrict__ st, const float* __restrict__ b2v,
    const long long* __restrict__ idx, const float* __restrict__ gx,
    const float* __restrict__ x, __bf16* __restrict__ x2g,
    float* __restrict__ sumM, float* __restrict__ sqM) {
    __shared__ __bf16 Ubuf[16 * 40];
    __shared__ float Rbuf[16 * 257];
    __shared__ __bf16 X2buf[16 * 264];
    __shared__ float red[16 * 16];
    __shared__ float rowStat[16];
    __shared__ float sLoc[C], qLoc[C];
    const int tid = threadIdx.x, lane = tid & 31, wv = tid >> 5;
    const int row0 = blockIdx.x * 16;
    const float* sb = st + SB;
    const float* bb = st + BB;
    for (int i = tid; i < C; i += 256) { sLoc[i] = 0.f; qLoc[i] = 0.f; }
    // U = relu(bn(t))  (16 x 32, bf16)
    for (int i = tid; i < 16 * H; i += 256) {
        int r = i >> 5, k = i & 31;
        float v = t[(size_t)(row0 + r) * H + k] * sb[k] + bb[k];
        Ubuf[r * 40 + k] = (__bf16)fmaxf(v, 0.0f);
    }
    __syncthreads();
    // GEMM2: R = U @ W2^T + b2  (K = 32 -> one WMMA per tile)
    {
        const int mloc = lane & 15, kbase = (lane >> 4) << 3;
        v8bf lo = *(const v8bf*)(Ubuf + mloc * 40 + kbase);
        v8bf hi = *(const v8bf*)(Ubuf + mloc * 40 + kbase + 16);
        v16bf a = cat16(lo, hi);
#pragma unroll
        for (int i = 0; i < 2; ++i) {
            int ct = wv * 2 + i;
            v16bf b = *(const v16bf*)(pk + W2P_H + (size_t)ct * 512 + lane * 16);
            v8f acc = {};
            acc = __builtin_amdgcn_wmma_f32_16x16x32_bf16(false, a, false, b,
                                                          (short)0, acc, false, false);
            int n = ct * 16 + (lane & 15);
            float bias = b2v[n];
#pragma unroll
            for (int j = 0; j < 8; ++j) {
                int m = j + ((lane >> 4) << 3);
                Rbuf[m * 257 + n] = acc[j] + bias;
            }
        }
    }
    __syncthreads();
    // row softmax over C (16 threads per row)
    {
        int r = tid >> 4, j = tid & 15;
        float mx = -3.4e38f;
        for (int c = j; c < C; c += 16) mx = fmaxf(mx, Rbuf[r * 257 + c]);
        red[r * 16 + j] = mx;
        __syncthreads();
        if (j == 0) {
            float m2 = red[r * 16];
            for (int u = 1; u < 16; ++u) m2 = fmaxf(m2, red[r * 16 + u]);
            rowStat[r] = m2;
        }
        __syncthreads();
        float mrow = rowStat[r];
        float s = 0.f;
        for (int c = j; c < C; c += 16) {
            float e = __expf(Rbuf[r * 257 + c] - mrow);
            Rbuf[r * 257 + c] = e;
            s += e;
        }
        red[r * 16 + j] = s;
        __syncthreads();
        if (j == 0) {
            float s2 = 0.f;
            for (int u = 0; u < 16; ++u) s2 += red[r * 16 + u];
            rowStat[r] = 1.0f / s2;
        }
    }
    __syncthreads();
    // X2 = x + softmax * gx[idx]; keep in LDS and stream to global (bf16)
    {
        int c = tid;
        for (int r = 0; r < 16; ++r) {
            long long g = idx[row0 + r];
            float aug = gx[(size_t)g * C + c];
            float xv = x[(size_t)(row0 + r) * C + c];
            float sw = Rbuf[r * 257 + c] * rowStat[r];
            __bf16 hv = (__bf16)(xv + sw * aug);
            X2buf[r * 264 + c] = hv;
            x2g[(size_t)(row0 + r) * C + c] = hv;
        }
    }
    __syncthreads();
    // GEMM3 (stats only): accumulate per-channel sum/sumsq of z = X2 @ Wlin^T
    {
        const int mloc = lane & 15, kbase = (lane >> 4) << 3;
#pragma unroll
        for (int i = 0; i < 2; ++i) {
            int ct = wv * 2 + i;
            v8f acc = {};
            for (int ks = 0; ks < C / 32; ++ks) {
                v8bf lo = *(const v8bf*)(X2buf + mloc * 264 + ks * 32 + kbase);
                v8bf hi = *(const v8bf*)(X2buf + mloc * 264 + ks * 32 + kbase + 16);
                v16bf a = cat16(lo, hi);
                v16bf b = *(const v16bf*)(pk + WLP_H + ((size_t)(ct * 8 + ks)) * 512 + lane * 16);
                acc = __builtin_amdgcn_wmma_f32_16x16x32_bf16(false, a, false, b,
                                                              (short)0, acc, false, false);
            }
            int n = ct * 16 + (lane & 15);
            float s = 0.f, q = 0.f;
#pragma unroll
            for (int j = 0; j < 8; ++j) {
                float v = acc[j];
                s += v;
                q += v * v;
            }
            atomicAdd(&sLoc[n], s);
            atomicAdd(&qLoc[n], q);
        }
    }
    __syncthreads();
    if (tid < C) { atomicAdd(&sumM[tid], sLoc[tid]); atomicAdd(&sqM[tid], qLoc[tid]); }
}

// ---------------- kernel 4: out = relu(bn(x2 @ Wlin^T))  (WMMA recompute) ----------------
__global__ void __launch_bounds__(256) k_out(const __bf16* __restrict__ x2,
                                             const __bf16* __restrict__ pk,
                                             const float* __restrict__ st,
                                             float* __restrict__ out) {
    __shared__ __bf16 X2buf[16 * 264];
    const int tid = threadIdx.x, lane = tid & 31, wv = tid >> 5;
    const int row0 = blockIdx.x * 16;
    // stage the 16x256 row block once (coalesced 32B per thread)
    {
        int r = tid >> 4, seg = tid & 15;
        const v8bf* src = (const v8bf*)(x2 + (size_t)(row0 + r) * C + seg * 16);
        v8bf a0 = src[0];
        v8bf a1 = src[1];
        *(v8bf*)(X2buf + r * 264 + seg * 16) = a0;
        *(v8bf*)(X2buf + r * 264 + seg * 16 + 8) = a1;
    }
    __syncthreads();
    const float* sm = st + SM;
    const float* bm = st + BM;
    const int mloc = lane & 15, kbase = (lane >> 4) << 3;
#pragma unroll
    for (int i = 0; i < 2; ++i) {
        int ct = wv * 2 + i;
        v8f acc = {};
#pragma unroll
        for (int ks = 0; ks < C / 32; ++ks) {
            v8bf lo = *(const v8bf*)(X2buf + mloc * 264 + ks * 32 + kbase);
            v8bf hi = *(const v8bf*)(X2buf + mloc * 264 + ks * 32 + kbase + 16);
            v16bf a = cat16(lo, hi);
            v16bf b = *(const v16bf*)(pk + WLP_H + ((size_t)(ct * 8 + ks)) * 512 + lane * 16);
            acc = __builtin_amdgcn_wmma_f32_16x16x32_bf16(false, a, false, b,
                                                          (short)0, acc, false, false);
        }
        int n = ct * 16 + (lane & 15);
        float s = sm[n], sh = bm[n];
#pragma unroll
        for (int j = 0; j < 8; ++j) {
            int m = j + ((lane >> 4) << 3);
            out[(size_t)(row0 + m) * C + n] = fmaxf(acc[j] * s + sh, 0.0f);
        }
    }
}

extern "C" void kernel_launch(void* const* d_in, const int* in_sizes, int n_in,
                              void* d_out, int out_size, void* d_ws, size_t ws_size,
                              hipStream_t stream) {
    (void)in_sizes; (void)n_in; (void)out_size; (void)ws_size;
    const float* x      = (const float*)d_in[1];
    const float* gx     = (const float*)d_in[2];
    const long long* idx = (const long long*)d_in[3];
    const float* Wlin   = (const float*)d_in[4];
    const float* g_main = (const float*)d_in[5];
    const float* b_main = (const float*)d_in[6];
    const float* g_a    = (const float*)d_in[7];
    const float* b_a    = (const float*)d_in[8];
    const float* W1     = (const float*)d_in[9];
    const float* b1     = (const float*)d_in[10];
    const float* g_b    = (const float*)d_in[11];
    const float* b_b    = (const float*)d_in[12];
    const float* W2     = (const float*)d_in[13];
    const float* b2     = (const float*)d_in[14];

    char* ws = (char*)d_ws;
    __bf16* rel = (__bf16*)(ws + REL_OFF);
    __bf16* x2  = (__bf16*)(ws + X2_OFF);
    float* t    = (float*)(ws + T_OFF);
    __bf16* pk  = (__bf16*)(ws + PK_OFF);
    float* st   = (float*)(ws + STAT_OFF);
    float* out  = (float*)d_out;

    int prepThreads = (int)PK_HALVES + NSTAT_ZERO;
    k_prep<<<(prepThreads + 255) / 256, 256, 0, stream>>>(W1, W2, Wlin, pk, st);
    k_rel_stats<<<N / 32, 256, 0, stream>>>(x, gx, idx, rel, st + SUMA, st + SQA);
    k_finalize<<<1, 256, 0, stream>>>(st + SUMA, st + SQA, g_a, b_a,
                                      st + SA, st + BA, C, 1.0f / N);
    k_gemm1<<<N / 64, 256, 0, stream>>>(rel, pk, st, b1, t, st + SUMB, st + SQB);
    k_finalize<<<1, 32, 0, stream>>>(st + SUMB, st + SQB, g_b, b_b,
                                     st + SB, st + BB, H, 1.0f / N);
    k_fused2<<<N / 16, 256, 0, stream>>>(t, pk, st, b2, idx, gx, x, x2,
                                         st + SUMM, st + SQM);
    k_finalize<<<1, 256, 0, stream>>>(st + SUMM, st + SQM, g_main, b_main,
                                      st + SM, st + BM, C, 1.0f / N);
    k_out<<<N / 16, 256, 0, stream>>>(x2, pk, st, out);
}
